// GAT_5050881540194
// MI455X (gfx1250) — matrix operand order repaired and checked by reading, not compiled, not run on hardware
//
#include <hip/hip_runtime.h>
#include <hip/hip_bf16.h>

typedef __attribute__((ext_vector_type(16))) _Float16 v16h;
typedef __attribute__((ext_vector_type(8)))  _Float16 v8h;
typedef __attribute__((ext_vector_type(8)))  float    v8f;

#define NHEAD 4
#define SLOPE_ATT 0.2f
#define SLOPE_ACT 0.1f
#define BN_EPS 1e-5f

// ---------------------------------------------------------------------------
// utility kernels
// ---------------------------------------------------------------------------
__global__ void k_zero_u32(unsigned int* __restrict__ p, long n) {
    long i = (long)blockIdx.x * blockDim.x + threadIdx.x;
    if (i < n) p[i] = 0u;
}

__device__ __forceinline__ void atomic_max_f32(float* addr, float val) {
    if (val >= 0.0f) atomicMax((int*)addr, __float_as_int(val));
    else             atomicMin((unsigned int*)addr, (unsigned int)__float_as_int(val));
}

// ---------------------------------------------------------------------------
// graph build: edges + self loops, then CSR by dst (counting sort)
// ---------------------------------------------------------------------------
__global__ void k_build_edges(const long long* __restrict__ adj, int E, int N,
                              int* __restrict__ src, int* __restrict__ dst) {
    int i = blockIdx.x * blockDim.x + threadIdx.x;
    int Etot = E + N;
    if (i >= Etot) return;
    if (i < E) { src[i] = (int)adj[i]; dst[i] = (int)adj[(long)E + i]; }
    else       { src[i] = i - E;       dst[i] = i - E; }
}

__global__ void k_degree(const int* __restrict__ dst, int Etot, int* __restrict__ deg) {
    int i = blockIdx.x * blockDim.x + threadIdx.x;
    if (i < Etot) atomicAdd(&deg[dst[i]], 1);
}

__global__ void k_scan_block(const int* __restrict__ deg, int n,
                             int* __restrict__ excl, int* __restrict__ bsum) {
    __shared__ int sh[256];
    int i = blockIdx.x * 256 + threadIdx.x;
    int v = (i < n) ? deg[i] : 0;
    sh[threadIdx.x] = v;
    __syncthreads();
    for (int off = 1; off < 256; off <<= 1) {
        int t = 0;
        if ((int)threadIdx.x >= off) t = sh[threadIdx.x - off];
        __syncthreads();
        if ((int)threadIdx.x >= off) sh[threadIdx.x] += t;
        __syncthreads();
    }
    if (i < n) excl[i] = sh[threadIdx.x] - v;      // exclusive within block
    if (threadIdx.x == 255) bsum[blockIdx.x] = sh[255];
}

__global__ void k_scan_serial(int* __restrict__ bsum, int nb) {
    if (threadIdx.x == 0 && blockIdx.x == 0) {
        int acc = 0;
        for (int b = 0; b < nb; b++) { int t = bsum[b]; bsum[b] = acc; acc += t; }
    }
}

__global__ void k_scan_add(const int* __restrict__ excl, const int* __restrict__ bsum,
                           int n, int Etot, int* __restrict__ rowptr) {
    int i = blockIdx.x * 256 + threadIdx.x;
    if (i < n) rowptr[i] = excl[i] + bsum[blockIdx.x];
    if (i == 0) rowptr[n] = Etot;
}

__global__ void k_scatter(const int* __restrict__ dst, int Etot,
                          const int* __restrict__ rowptr, int* __restrict__ cursor,
                          int* __restrict__ perm) {
    int e = blockIdx.x * blockDim.x + threadIdx.x;
    if (e >= Etot) return;
    int d = dst[e];
    int p = atomicAdd(&cursor[d], 1);
    perm[rowptr[d] + p] = e;
}

// ---------------------------------------------------------------------------
// precision conversion
// ---------------------------------------------------------------------------
__global__ void k_wt_f16(const float* __restrict__ W, int K, int Nt,
                         _Float16* __restrict__ Wt) {     // W[K,Nt] -> Wt[Nt,K] f16
    int i = blockIdx.x * blockDim.x + threadIdx.x;
    if (i >= K * Nt) return;
    int k = i / Nt, n = i % Nt;
    Wt[(size_t)n * K + k] = (_Float16)W[i];
}

__global__ void k_to_f16_pad(const float* __restrict__ x, int n, int Mp, int C,
                             _Float16* __restrict__ y) {
    long i = (long)blockIdx.x * blockDim.x + threadIdx.x;
    if (i >= (long)Mp * C) return;
    int r = (int)(i / C);
    y[i] = (r < n) ? (_Float16)x[i] : (_Float16)0.0f;
}

// ---------------------------------------------------------------------------
// WMMA GEMM: C[Mp,Ntot] = A[Mp,K](f16,row-major) x Bt[Ntot,K](f16, = W^T)
// one wave per 16x(16*NTC) output strip; NTC chosen at compile time so the
// inner loop is branch-free: pure load_b128 -> v_wmma.
// ---------------------------------------------------------------------------
template <int NTC>
__global__ void k_gemm_wmma(const _Float16* __restrict__ A,
                            const _Float16* __restrict__ Bt,
                            float* __restrict__ C, int Mp, int K, int Ntot) {
    const int lane  = threadIdx.x;           // 0..31
    const int tileM = blockIdx.x;
    const int nbase = blockIdx.y * NTC;      // first N-tile of this wave
    const int hi    = lane >> 4;             // 0 | 1 (half-wave)
    const int hiA   = hi << 3;               // 0 | 8, unused (A also uses hiB)
    const int hiB   = hi << 4;               // B K-offset 0 | 16
    const int ncl   = lane & 15;
    const int mrow  = (tileM << 4) + ncl;

    const _Float16* __restrict__ arow = A + (size_t)mrow * K;
    const _Float16* brow[NTC];
#pragma unroll
    for (int t = 0; t < NTC; t++)
        brow[t] = Bt + (size_t)(((nbase + t) << 4) + ncl) * K + hiB;

    v8f acc[NTC];
#pragma unroll
    for (int t = 0; t < NTC; t++) acc[t] = (v8f){};

    for (int k0 = 0; k0 < K; k0 += 32) {
        union { v16h v; v8h h[2]; } a;
        a.h[0] = *(const v8h*)(arow + k0 + hiB);
        a.h[1] = *(const v8h*)(arow + k0 + hiB + 8);
#pragma unroll
        for (int t = 0; t < NTC; t++) {
            union { v16h v; v8h h[2]; } b;
            const _Float16* bk = brow[t] + k0;
            b.h[0] = *(const v8h*)bk;
            b.h[1] = *(const v8h*)(bk + 8);
            acc[t] = __builtin_amdgcn_wmma_f32_16x16x32_f16(false, a.v, false, b.v,
                                                            (short)0, acc[t], false, false);
        }
    }

    const int m0 = (tileM << 4) + (hi << 3);
#pragma unroll
    for (int t = 0; t < NTC; t++) {
        int nc = ((nbase + t) << 4) + ncl;
#pragma unroll
        for (int r = 0; r < 8; r++)
            C[(size_t)(m0 + r) * Ntot + nc] = acc[t][r];
    }
}

// ---------------------------------------------------------------------------
// attention: a_src/a_dst = <h[n,hh,:], att[hh,:]>
// ---------------------------------------------------------------------------
__global__ void k_att_proj(const float* __restrict__ h, int n, int Ch, int Ctot,
                           const float* __restrict__ att_s, const float* __restrict__ att_d,
                           float* __restrict__ a_s, float* __restrict__ a_d) {
    int i = blockIdx.x * blockDim.x + threadIdx.x;
    if (i >= n * NHEAD) return;
    int node = i >> 2, hh = i & 3;
    const float* hp = h + (size_t)node * Ctot + hh * Ch;
    const float* ws = att_s + hh * Ch;
    const float* wd = att_d + hh * Ch;
    float s1 = 0.f, s2 = 0.f;
    for (int c = 0; c < Ch; c++) { float v = hp[c]; s1 += v * ws[c]; s2 += v * wd[c]; }
    a_s[i] = s1; a_d[i] = s2;
}

__global__ void k_init_ms(float* __restrict__ m, float* __restrict__ s, int n) {
    int i = blockIdx.x * blockDim.x + threadIdx.x;
    if (i < n) { m[i] = -INFINITY; s[i] = 0.f; }
}

__global__ void k_edge_max(const int* __restrict__ src, const int* __restrict__ dst, int Etot,
                           const float* __restrict__ a_s, const float* __restrict__ a_d,
                           float* __restrict__ m) {
    int i = blockIdx.x * blockDim.x + threadIdx.x;
    if (i >= Etot * NHEAD) return;
    int e = i >> 2, hh = i & 3;
    float v = a_s[src[e] * NHEAD + hh] + a_d[dst[e] * NHEAD + hh];
    v = v > 0.f ? v : SLOPE_ATT * v;
    atomic_max_f32(&m[dst[e] * NHEAD + hh], v);
}

__global__ void k_edge_exp(const int* __restrict__ src, const int* __restrict__ dst, int Etot,
                           const float* __restrict__ a_s, const float* __restrict__ a_d,
                           const float* __restrict__ m, float* __restrict__ ee,
                           float* __restrict__ s) {
    int i = blockIdx.x * blockDim.x + threadIdx.x;
    if (i >= Etot * NHEAD) return;
    int e = i >> 2, hh = i & 3;
    int dn = dst[e];
    float v = a_s[src[e] * NHEAD + hh] + a_d[dn * NHEAD + hh];
    v = v > 0.f ? v : SLOPE_ATT * v;
    float x = __expf(v - m[dn * NHEAD + hh]);
    ee[i] = x;
    atomicAdd(&s[dn * NHEAD + hh], x);
}

// ---------------------------------------------------------------------------
// aggregation (CSR, no atomics): one block per dst node, one thread per channel
// ---------------------------------------------------------------------------
__global__ void k_agg_concat(const int* __restrict__ rowptr, const int* __restrict__ perm,
                             const int* __restrict__ src,
                             const float* __restrict__ h, const float* __restrict__ ee,
                             const float* __restrict__ s, const float* __restrict__ bias,
                             float* __restrict__ out, int Ctot, int Ch) {
    int node = blockIdx.x;
    int c = threadIdx.x;
    int hh = c / Ch;
    float inv = 1.0f / (s[node * NHEAD + hh] + 1e-16f);
    int beg = rowptr[node], end = rowptr[node + 1];
    float acc = 0.f;
    for (int i = beg; i < end; i++) {
        int e = perm[i];
        if (i + 1 < end)
            __builtin_prefetch(&h[(size_t)src[perm[i + 1]] * Ctot + c], 0, 0);
        acc += h[(size_t)src[e] * Ctot + c] * ee[e * NHEAD + hh];
    }
    out[(size_t)node * Ctot + c] = acc * inv + bias[c];
}

__global__ void k_agg_mean(const int* __restrict__ rowptr, const int* __restrict__ perm,
                           const int* __restrict__ src,
                           const float* __restrict__ h, const float* __restrict__ ee,
                           const float* __restrict__ s, const float* __restrict__ bias,
                           float* __restrict__ out) {           // Ctot=160, Ch=40 -> out[.,40]
    __shared__ float sh[160];
    int node = blockIdx.x;
    int c = threadIdx.x;                                        // 0..159
    int hh = c / 40;
    float inv = 1.0f / (s[node * NHEAD + hh] + 1e-16f);
    int beg = rowptr[node], end = rowptr[node + 1];
    float acc = 0.f;
    for (int i = beg; i < end; i++) {
        int e = perm[i];
        if (i + 1 < end)
            __builtin_prefetch(&h[(size_t)src[perm[i + 1]] * 160 + c], 0, 0);
        acc += h[(size_t)src[e] * 160 + c] * ee[e * NHEAD + hh];
    }
    sh[c] = acc * inv;
    __syncthreads();
    if (c < 40)
        out[(size_t)node * 40 + c] =
            (sh[c] + sh[c + 40] + sh[c + 80] + sh[c + 120]) * 0.25f + bias[c];
}

// ---------------------------------------------------------------------------
// batchnorm (biased var over N) + leakyrelu(0.1) fused into f16 GEMM-A producer
// ---------------------------------------------------------------------------
__global__ void k_bn_stats(const float* __restrict__ x, int n, int C,
                           float* __restrict__ sum, float* __restrict__ sumsq) {
    int c = threadIdx.x;                 // blockDim == C
    float s1 = 0.f, s2 = 0.f;
    for (int r = blockIdx.x; r < n; r += gridDim.x) {
        float v = x[(size_t)r * C + c];
        s1 += v; s2 += v * v;
    }
    atomicAdd(&sum[c], s1);
    atomicAdd(&sumsq[c], s2);
}

__global__ void k_bn_final(const float* __restrict__ sum, const float* __restrict__ sumsq,
                           int n, const float* __restrict__ g, const float* __restrict__ be,
                           float* __restrict__ scale, float* __restrict__ shift, int C) {
    int c = blockIdx.x * blockDim.x + threadIdx.x;
    if (c >= C) return;
    float mu  = sum[c] / (float)n;
    float var = sumsq[c] / (float)n - mu * mu;
    float sc  = g[c] * rsqrtf(var + BN_EPS);
    scale[c] = sc;
    shift[c] = be[c] - mu * sc;
}

__global__ void k_bn_apply_f16(const float* __restrict__ x, int n, int Mp, int C,
                               const float* __restrict__ scale, const float* __restrict__ shift,
                               _Float16* __restrict__ y) {
    long i = (long)blockIdx.x * blockDim.x + threadIdx.x;
    if (i >= (long)Mp * C) return;
    int r = (int)(i / C), c = (int)(i % C);
    float v = 0.f;
    if (r < n) {
        v = x[i] * scale[c] + shift[c];
        v = v > 0.f ? v : SLOPE_ACT * v;
    }
    y[i] = (_Float16)v;
}

__global__ void k_log_softmax40(const float* __restrict__ x, float* __restrict__ out, int n) {
    int node = blockIdx.x * blockDim.x + threadIdx.x;
    if (node >= n) return;
    const float* p = x + (size_t)node * 40;
    float mx = -INFINITY;
    for (int c = 0; c < 40; c++) mx = fmaxf(mx, p[c]);
    float ss = 0.f;
    for (int c = 0; c < 40; c++) ss += __expf(p[c] - mx);
    float lse = mx + __logf(ss);
    for (int c = 0; c < 40; c++) out[(size_t)node * 40 + c] = p[c] - lse;
}

// ---------------------------------------------------------------------------
// host orchestration
// ---------------------------------------------------------------------------
extern "C" void kernel_launch(void* const* d_in, const int* in_sizes, int n_in,
                              void* d_out, int out_size, void* d_ws, size_t ws_size,
                              hipStream_t stream) {
    const float*     x   = (const float*)d_in[0];
    const long long* adj = (const long long*)d_in[1];
    const float* W0  = (const float*)d_in[2];
    const float* as0 = (const float*)d_in[3];
    const float* ad0 = (const float*)d_in[4];
    const float* b0  = (const float*)d_in[5];
    const float* W1  = (const float*)d_in[6];
    const float* as1 = (const float*)d_in[7];
    const float* ad1 = (const float*)d_in[8];
    const float* b1  = (const float*)d_in[9];
    const float* W2  = (const float*)d_in[10];
    const float* as2 = (const float*)d_in[11];
    const float* ad2 = (const float*)d_in[12];
    const float* b2  = (const float*)d_in[13];
    const float* g0  = (const float*)d_in[14];
    const float* be0 = (const float*)d_in[15];
    const float* g1  = (const float*)d_in[16];
    const float* be1 = (const float*)d_in[17];

    const int N    = in_sizes[0] / 128;
    const int E    = in_sizes[1] / 2;
    const int Etot = E + N;
    const int Mp   = ((N + 15) / 16) * 16;
    const int nb   = (N + 255) / 256;

    // --- workspace bump allocator ---
    char* base = (char*)d_ws;
    size_t off = 0;
    auto alloc = [&](size_t bytes) -> void* {
        void* p = base + off;
        off = (off + bytes + 255) & ~(size_t)255;
        return p;
    };
    int* src32   = (int*)alloc((size_t)Etot * 4);
    int* dst32   = (int*)alloc((size_t)Etot * 4);
    int* deg     = (int*)alloc((size_t)N * 4);
    int* excl    = (int*)alloc((size_t)N * 4);
    int* bsum    = (int*)alloc((size_t)nb * 4);
    int* rowptr  = (int*)alloc((size_t)(N + 1) * 4);
    int* cursor  = (int*)alloc((size_t)N * 4);
    int* perm    = (int*)alloc((size_t)Etot * 4);
    _Float16* A16  = (_Float16*)alloc((size_t)Mp * 256 * 2);
    _Float16* Wt16 = (_Float16*)alloc((size_t)256 * 256 * 2);
    float* Hbuf  = (float*)alloc((size_t)Mp * 256 * 4);
    float* Obuf  = (float*)alloc((size_t)Mp * 256 * 4);
    float* a_s   = (float*)alloc((size_t)N * NHEAD * 4);
    float* a_d   = (float*)alloc((size_t)N * NHEAD * 4);
    float* m_    = (float*)alloc((size_t)N * NHEAD * 4);
    float* s_    = (float*)alloc((size_t)N * NHEAD * 4);
    float* ee    = (float*)alloc((size_t)Etot * NHEAD * 4);
    float* bnsum = (float*)alloc(256 * 4);
    float* bnsq  = (float*)alloc(256 * 4);
    float* bnsc  = (float*)alloc(256 * 4);
    float* bnsh  = (float*)alloc(256 * 4);

    const int TB = 256;
    const int gE   = (Etot + TB - 1) / TB;
    const int gEH  = (Etot * NHEAD + TB - 1) / TB;
    const int gNH  = (N * NHEAD + TB - 1) / TB;

    // --- build graph CSR (by dst) once; reused by all 3 layers ---
    k_zero_u32<<<(N + TB - 1) / TB, TB, 0, stream>>>((unsigned*)deg, N);
    k_build_edges<<<gE, TB, 0, stream>>>(adj, E, N, src32, dst32);
    k_degree<<<gE, TB, 0, stream>>>(dst32, Etot, deg);
    k_scan_block<<<nb, 256, 0, stream>>>(deg, N, excl, bsum);
    k_scan_serial<<<1, 1, 0, stream>>>(bsum, nb);
    k_scan_add<<<nb, 256, 0, stream>>>(excl, bsum, N, Etot, rowptr);
    k_zero_u32<<<(N + TB - 1) / TB, TB, 0, stream>>>((unsigned*)cursor, N);
    k_scatter<<<gE, TB, 0, stream>>>(dst32, Etot, rowptr, cursor, perm);

    // shared edge-softmax pipeline for one layer
    auto edge_softmax = [&](int Ch, int Ctot,
                            const float* atts, const float* attd) {
        k_att_proj<<<gNH, TB, 0, stream>>>(Hbuf, N, Ch, Ctot, atts, attd, a_s, a_d);
        k_init_ms<<<gNH, TB, 0, stream>>>(m_, s_, N * NHEAD);
        k_edge_max<<<gEH, TB, 0, stream>>>(src32, dst32, Etot, a_s, a_d, m_);
        k_edge_exp<<<gEH, TB, 0, stream>>>(src32, dst32, Etot, a_s, a_d, m_, ee, s_);
    };
    auto bn_act = [&](const float* g, const float* be) {  // Obuf -> A16 (f16, padded)
        k_zero_u32<<<2, 256, 0, stream>>>((unsigned*)bnsum, 256);
        k_zero_u32<<<2, 256, 0, stream>>>((unsigned*)bnsq, 256);
        k_bn_stats<<<512, 256, 0, stream>>>(Obuf, N, 256, bnsum, bnsq);
        k_bn_final<<<1, 256, 0, stream>>>(bnsum, bnsq, N, g, be, bnsc, bnsh, 256);
        long tot = (long)Mp * 256;
        k_bn_apply_f16<<<(unsigned)((tot + TB - 1) / TB), TB, 0, stream>>>(
            Obuf, N, Mp, 256, bnsc, bnsh, A16);
    };

    // --- layer 0: 128 -> 4x64 concat ---
    {
        long tot = (long)Mp * 128;
        k_to_f16_pad<<<(unsigned)((tot + TB - 1) / TB), TB, 0, stream>>>(x, N, Mp, 128, A16);
        k_wt_f16<<<(128 * 256 + TB - 1) / TB, TB, 0, stream>>>(W0, 128, 256, Wt16);
        k_gemm_wmma<4><<<dim3(Mp / 16, 256 / 64), 32, 0, stream>>>(
            A16, Wt16, Hbuf, Mp, 128, 256);
        edge_softmax(64, 256, as0, ad0);
        k_agg_concat<<<N, 256, 0, stream>>>(rowptr, perm, src32, Hbuf, ee, s_, b0,
                                            Obuf, 256, 64);
        bn_act(g0, be0);
    }

    // --- layer 1: 256 -> 4x64 concat ---
    {
        k_wt_f16<<<(256 * 256 + TB - 1) / TB, TB, 0, stream>>>(W1, 256, 256, Wt16);
        k_gemm_wmma<4><<<dim3(Mp / 16, 256 / 64), 32, 0, stream>>>(
            A16, Wt16, Hbuf, Mp, 256, 256);
        edge_softmax(64, 256, as1, ad1);
        k_agg_concat<<<N, 256, 0, stream>>>(rowptr, perm, src32, Hbuf, ee, s_, b1,
                                            Obuf, 256, 64);
        bn_act(g1, be1);
    }

    // --- layer 2: 256 -> 4x40 mean, then log_softmax ---
    {
        k_wt_f16<<<(256 * 160 + TB - 1) / TB, TB, 0, stream>>>(W2, 256, 160, Wt16);
        k_gemm_wmma<2><<<dim3(Mp / 16, 160 / 32), 32, 0, stream>>>(
            A16, Wt16, Hbuf, Mp, 256, 160);
        edge_softmax(40, 160, as2, ad2);
        k_agg_mean<<<N, 160, 0, stream>>>(rowptr, perm, src32, Hbuf, ee, s_, b2, Obuf);
        k_log_softmax40<<<(N + TB - 1) / TB, TB, 0, stream>>>(Obuf, (float*)d_out, N);
    }
    (void)n_in; (void)out_size; (void)ws_size;
}
